// MultiScaleTemporalAttention_86577950753236
// MI455X (gfx1250) — matrix operand order repaired and checked by reading, not compiled, not run on hardware
//
#include <hip/hip_runtime.h>

typedef __attribute__((ext_vector_type(16))) _Float16 v16h;
typedef __attribute__((ext_vector_type(8)))  _Float16 v8h;
typedef __attribute__((ext_vector_type(8)))  float    v8f;

#define SEQ   4096
#define HID   512
#define NHEAD 8
#define HD    64

__device__ __forceinline__ v8f wmma_ff(v16h a, v16h b, v8f c) {
  // D = A(16x32 f16) * B(32x16 f16) + C(16x16 f32)
  return __builtin_amdgcn_wmma_f32_16x16x32_f16(false, a, false, b, (short)0, c,
                                                false, false);
}

// ---- A fragment: 16x32 f16, rows m0.., K chunk k0.. (row-major source) ----
// ISA layout: lane&15 = row M; half = lane>>4; elem[2v+j] -> K = (v<4?0:16)+8*half+2*(v&3)+j
__device__ __forceinline__ v16h load_a_frag(const _Float16* base, long row_stride,
                                            int m0, int k0) {
  const int lane = threadIdx.x & 31;
  const _Float16* p = base + (long)(m0 + (lane & 15)) * row_stride + k0 + 8 * (lane >> 4);
  v8h lo = *(const v8h*)(p);
  v8h hi = *(const v8h*)(p + 16);
  v16h a;
#pragma unroll
  for (int i = 0; i < 8; ++i) { a[i] = lo[i]; a[8 + i] = hi[i]; }
  return a;
}

// ---- B fragment: logical B(32x16) = X^T, X row-major with rows = N, cols = K ----
// ISA layout: lane&15 = col N; half = lane>>4; elem[e] -> K = k0 + (e<8?0:16) + 8*half + (e&7)
__device__ __forceinline__ v16h load_bT_frag(const _Float16* X, long ld,
                                             int n0, int k0) {
  const int lane = threadIdx.x & 31;
  const _Float16* p = X + (long)(n0 + (lane & 15)) * ld + k0 + 8 * (lane >> 4);
  v8h lo = *(const v8h*)(p);
  v8h hi = *(const v8h*)(p + 16);
  v16h b;
#pragma unroll
  for (int i = 0; i < 8; ++i) { b[i] = lo[i]; b[8 + i] = hi[i]; }
  return b;
}

// ---- async global->LDS 16B copy (CDNA5 ASYNCcnt path) via inline asm ----
// ISA 15.18.3: GLOBAL_LOAD_ASYNC_TO_LDS_B128, VDST = LDS address VGPR, GV mode (SADDR=off).
// LDS aperture rule: generic addr low 32 bits = LDS byte address.
__device__ __forceinline__ void async_g2l_b128(const _Float16* g, _Float16* l) {
  const unsigned lds_off = (unsigned)(size_t)l;
  asm volatile("global_load_async_to_lds_b128 %0, %1, off"
               :: "v"(lds_off), "v"(g) : "memory");
}

__device__ __forceinline__ void wait_async0() {
  asm volatile("s_wait_asynccnt 0x0" ::: "memory");
}

// ---- 16-lane butterfly max via DPP (pure VALU, no DS waits) ----
__device__ __forceinline__ float rowmax16(float x) {
#define DPP_MAX_STEP(ctrl)                                                       \
  {                                                                              \
    int t_ = __builtin_amdgcn_update_dpp(0, __builtin_bit_cast(int, x), (ctrl),  \
                                         0xF, 0xF, true);                        \
    x = fmaxf(x, __builtin_bit_cast(float, t_));                                 \
  }
  DPP_MAX_STEP(0xB1)   // quad_perm(1,0,3,2)  : xor 1
  DPP_MAX_STEP(0x4E)   // quad_perm(2,3,0,1)  : xor 2
  DPP_MAX_STEP(0x141)  // row_half_mirror     : combine 4-groups
  DPP_MAX_STEP(0x140)  // row_mirror          : combine 8-groups (within 16 lanes)
#undef DPP_MAX_STEP
  return x;
}

// ---------------- elementwise helpers ----------------
__global__ __launch_bounds__(256) void f32_to_f16_kernel(const float* __restrict__ s,
                                                         _Float16* __restrict__ d, int n) {
  int i = blockIdx.x * blockDim.x + threadIdx.x;
  if (i < n) d[i] = (_Float16)s[i];
}

// Convert 512x512 f32 W[k][n] -> f16 Wt[n][k] (one-time; makes B-frags contiguous)
__global__ __launch_bounds__(256) void f32_to_f16_t_kernel(const float* __restrict__ s,
                                                           _Float16* __restrict__ d) {
  int id = blockIdx.x * blockDim.x + threadIdx.x;
  if (id >= HID * HID) return;
  int k = id >> 9, n = id & 511;
  d[(long)n * HID + k] = (_Float16)s[id];
}

__global__ __launch_bounds__(256) void replicate_rows_kernel(const _Float16* __restrict__ y,
                                                             _Float16* __restrict__ stacked,
                                                             int scale, int sidx) {
  int id = blockIdx.x * blockDim.x + threadIdx.x;
  if (id >= SEQ * HID) return;
  int t = id >> 9, c = id & 511;
  stacked[((long)t * 4 + sidx) * HID + c] = y[(long)(t / scale) * HID + c];
}

__global__ __launch_bounds__(256) void mean4_kernel(const _Float16* __restrict__ X,
                                                    _Float16* __restrict__ Y) {
  int id = blockIdx.x * blockDim.x + threadIdx.x;
  if (id >= SEQ * HID) return;
  int t = id >> 9, c = id & 511;
  const _Float16* p = X + (long)t * 4 * HID + c;
  float s = (float)p[0] + (float)p[HID] + (float)p[2 * HID] + (float)p[3 * HID];
  Y[id] = (_Float16)(0.25f * s);
}

// ---------------- WMMA GEMM: C[M x 512] = A[M x 512] @ W[512 x 512] + bias ----------------
// 4 waves / block; each wave owns a 16x64 tile of a 64x64 block tile. The 64x32 B tile is
// double-buffered in LDS via global_load_async_to_lds_b128 (shared by all 4 waves).
// Wt is the TRANSPOSED f16 weight (Wt[n][k]).
// OMODE: 0 = f16 row-major out, 1 = f16 transposed out (Ct[n][m], ld=M), 2 = f32 out.
template <int OMODE>
__global__ __launch_bounds__(128) void gemm512_wmma(const _Float16* __restrict__ A, long a_stride,
                                                    const _Float16* __restrict__ Wt,
                                                    const float* __restrict__ bias,
                                                    void* __restrict__ Cout, int M) {
  __shared__ __align__(16) _Float16 Bs[2][64 * 32];
  const int tid  = threadIdx.x;
  const int lane = tid & 31;
  const int wave = tid >> 5;
  const int n0 = blockIdx.x * 64;
  const int m0 = blockIdx.y * 64 + wave * 16;

  // stage Wt[n0..n0+63][k0..k0+31] (4KB) into Bs[buf]: 256 x 16B chunks, 2 per thread
  auto stage = [&](int buf, int k0) {
#pragma unroll
    for (int r = 0; r < 2; ++r) {
      const int c  = tid + r * 128;
      const int nl = c >> 2;           // 0..63
      const int kc = (c & 3) * 8;      // 0,8,16,24 (halfs)
      async_g2l_b128(Wt + (long)(n0 + nl) * HID + k0 + kc, &Bs[buf][nl * 32 + kc]);
    }
  };

  stage(0, 0);
  wait_async0();
  __syncthreads();

  v8f acc[4] = {};
  for (int it = 0; it < 16; ++it) {
    const int k0 = it * 32;
    const int buf = it & 1;
    if (it + 1 < 16) stage(buf ^ 1, k0 + 32);          // prefetch overlaps WMMAs
    v16h a = load_a_frag(A, a_stride, m0, k0);
#pragma unroll
    for (int t = 0; t < 4; ++t)
      acc[t] = wmma_ff(a, load_bT_frag(&Bs[buf][0], 32, t * 16, 0), acc[t]);
    wait_async0();
    __syncthreads();
  }

  const int col = lane & 15;
  const int rb = (lane >> 4) * 8;
#pragma unroll
  for (int t = 0; t < 4; ++t) {
    const int n = n0 + t * 16 + col;
    const float bv = bias[n];
#pragma unroll
    for (int v = 0; v < 8; ++v) {
      const int row = m0 + rb + v;
      const float r = acc[t][v] + bv;
      if (OMODE == 0) ((_Float16*)Cout)[(long)row * HID + n] = (_Float16)r;
      if (OMODE == 1) ((_Float16*)Cout)[(long)n * M + row] = (_Float16)r;
      if (OMODE == 2) ((float*)Cout)[(long)row * HID + n] = r;
    }
  }
}

// ---------------- flash attention: one wave per (head, 16-query tile) ----------------
// Q,K row-major (L x 512); Vt transposed (512 x L) so P*V B-frags load contiguously.
// Row max: DPP butterfly. Row sum: extra P @ ones WMMA column (rescales like O).
__global__ __launch_bounds__(32) void flash_attn_wmma(const _Float16* __restrict__ Q,
                                                      const _Float16* __restrict__ K,
                                                      const _Float16* __restrict__ Vt,
                                                      _Float16* __restrict__ O, int L) {
  __shared__ __align__(16) _Float16 Pls[16 * 32];
  const int h = blockIdx.x;
  const int m0 = blockIdx.y * 16;
  const int lane = threadIdx.x & 31;
  const int d0 = h * HD;

  v16h qa0 = load_a_frag(Q, HID, m0, d0);
  v16h qa1 = load_a_frag(Q, HID, m0, d0 + 32);
  v16h ones;
#pragma unroll
  for (int i = 0; i < 16; ++i) {     // fold 1/sqrt(64) into Q (exact pow2 scale)
    qa0[i] = (_Float16)((float)qa0[i] * 0.125f);
    qa1[i] = (_Float16)((float)qa1[i] * 0.125f);
    ones[i] = (_Float16)1.0f;
  }

  float mrow[8];
#pragma unroll
  for (int v = 0; v < 8; ++v) mrow[v] = -3.0e38f;
  v8f o0 = {}, o1 = {}, o2 = {}, o3 = {}, osum = {};

  const int rb = (lane >> 4) * 8;   // C-layout row base
  const int col = lane & 15;        // C-layout column

  for (int kb = 0; kb < L; kb += 32) {
    v8f s0 = {}, s1 = {};
    s0 = wmma_ff(qa0, load_bT_frag(K, HID, kb,      d0     ), s0);
    s0 = wmma_ff(qa1, load_bT_frag(K, HID, kb,      d0 + 32), s0);
    s1 = wmma_ff(qa0, load_bT_frag(K, HID, kb + 16, d0     ), s1);
    s1 = wmma_ff(qa1, load_bT_frag(K, HID, kb + 16, d0 + 32), s1);

#pragma unroll
    for (int v = 0; v < 8; ++v) {
      const float mx = rowmax16(fmaxf(s0[v], s1[v]));
      const float mnew = fmaxf(mrow[v], mx);
      const float alpha = __expf(mrow[v] - mnew);
      mrow[v] = mnew;
      const float p0 = __expf(s0[v] - mnew);
      const float p1 = __expf(s1[v] - mnew);
      o0[v] *= alpha; o1[v] *= alpha; o2[v] *= alpha; o3[v] *= alpha; osum[v] *= alpha;
      Pls[(rb + v) * 32 + col]      = (_Float16)p0;
      Pls[(rb + v) * 32 + col + 16] = (_Float16)p1;
    }
    __builtin_amdgcn_wave_barrier();          // keep DS stores before DS loads
    v16h pa = load_a_frag(Pls, 32, 0, 0);     // re-layout P: C-frag -> A-frag via LDS
    __builtin_amdgcn_wave_barrier();          // keep loads before next-iter stores
    o0 = wmma_ff(pa, load_bT_frag(Vt, L, d0     , kb), o0);
    o1 = wmma_ff(pa, load_bT_frag(Vt, L, d0 + 16, kb), o1);
    o2 = wmma_ff(pa, load_bT_frag(Vt, L, d0 + 32, kb), o2);
    o3 = wmma_ff(pa, load_bT_frag(Vt, L, d0 + 48, kb), o3);
    osum = wmma_ff(pa, ones, osum);           // row sums: every lane gets the sum
  }

#pragma unroll
  for (int v = 0; v < 8; ++v) {
    const float inv = 1.0f / osum[v];
    const long base = (long)(m0 + rb + v) * HID + d0 + col;
    O[base]      = (_Float16)(o0[v] * inv);
    O[base + 16] = (_Float16)(o1[v] * inv);
    O[base + 32] = (_Float16)(o2[v] * inv);
    O[base + 48] = (_Float16)(o3[v] * inv);
  }
}

// ---------------- fusion attention: softmax over the 4 scales per (t, head, qi) ----------------
__global__ __launch_bounds__(256) void fusion_attn_kernel(const _Float16* __restrict__ Q,
                                                          const _Float16* __restrict__ K,
                                                          const _Float16* __restrict__ V,
                                                          _Float16* __restrict__ O) {
  const int id = blockIdx.x * blockDim.x + threadIdx.x;
  if (id >= SEQ * NHEAD * 4) return;
  const int qi = id & 3;
  const int h = (id >> 2) & 7;
  const int t = id >> 5;
  const long rb = (long)t * 4 * HID + h * HD;
  const _Float16* qp = Q + rb + (long)qi * HID;
  float s[4];
#pragma unroll
  for (int j = 0; j < 4; ++j) {
    const _Float16* kp = K + rb + (long)j * HID;
    float acc = 0.0f;
    for (int d = 0; d < HD; ++d) acc += (float)qp[d] * (float)kp[d];
    s[j] = acc * 0.125f;
  }
  const float mx = fmaxf(fmaxf(s[0], s[1]), fmaxf(s[2], s[3]));
  float p[4], den = 0.0f;
#pragma unroll
  for (int j = 0; j < 4; ++j) { p[j] = __expf(s[j] - mx); den += p[j]; }
  const float inv = 1.0f / den;
  _Float16* op = O + rb + (long)qi * HID;
  for (int d = 0; d < HD; ++d) {
    float acc = 0.0f;
#pragma unroll
    for (int j = 0; j < 4; ++j) acc += p[j] * (float)V[rb + (long)j * HID + d];
    op[d] = (_Float16)(acc * inv);
  }
}

// =======================================================================
extern "C" void kernel_launch(void* const* d_in, const int* in_sizes, int n_in,
                              void* d_out, int out_size, void* d_ws, size_t ws_size,
                              hipStream_t stream) {
  (void)n_in; (void)out_size; (void)ws_size;
  const int MAT = HID * HID;

  int pbase, xi;
  if (in_sizes[0] == SEQ * HID) { xi = 0; pbase = 2; }
  else                          { pbase = 0; xi = 50; }
  auto P = [&](int off) { return (const float*)d_in[pbase + off]; };
  const float* x_f32 = (const float*)d_in[xi];

  const float* Wout_f = P(0);  const float* bout = P(1);
  const float* fW[4]  = { P(4), P(2), P(5), P(3) };   // q, k, v, o
  const float* fb[4]  = { P(8), P(6), P(9), P(7) };
  const float *sW[4][5], *sb[4][5];                   // per scale: q,k,v,o,p
  for (int i = 0; i < 4; ++i) {
    int b = 10 + 10 * i;
    sW[i][0] = P(b + 3); sW[i][1] = P(b + 0); sW[i][2] = P(b + 4);
    sW[i][3] = P(b + 1); sW[i][4] = P(b + 2);
    sb[i][0] = P(b + 8); sb[i][1] = P(b + 5); sb[i][2] = P(b + 9);
    sb[i][3] = P(b + 6); sb[i][4] = P(b + 7);
  }

  char* ws = (char*)d_ws;
  size_t off = 0;
  auto take = [&](size_t bytes) {
    char* p = ws + off;
    off += (bytes + 255) & ~(size_t)255;
    return p;
  };
  const size_t MATB = (size_t)MAT * 2;
  const size_t ROWB = (size_t)SEQ * HID * 2;

  _Float16* Wout_h = (_Float16*)take(MATB);            // all weights stored TRANSPOSED f16
  _Float16* fWh[4]; for (int j = 0; j < 4; ++j) fWh[j] = (_Float16*)take(MATB);
  _Float16* sWh[4][5];
  for (int i = 0; i < 4; ++i) for (int j = 0; j < 5; ++j) sWh[i][j] = (_Float16*)take(MATB);

  _Float16* xh      = (_Float16*)take(ROWB);
  _Float16* Qh      = (_Float16*)take(ROWB);
  _Float16* Kh      = (_Float16*)take(ROWB);
  _Float16* Vt      = (_Float16*)take(ROWB);           // per-scale V, transposed (512 x L)
  _Float16* AOh     = (_Float16*)take(ROWB);
  _Float16* Yh      = (_Float16*)take(ROWB);
  _Float16* Ph      = (_Float16*)take(ROWB);
  _Float16* stacked = (_Float16*)take(ROWB * 4);
  _Float16* Qf      = (_Float16*)take(ROWB * 4);
  _Float16* Kf      = (_Float16*)take(ROWB * 4);
  _Float16* Vf      = (_Float16*)take(ROWB * 4);
  _Float16* AOf     = (_Float16*)take(ROWB * 4);
  _Float16* meanAO  = (_Float16*)take(ROWB);
  _Float16* T1      = (_Float16*)take(ROWB);

  const dim3 blk256(256);
  auto cvtT = [&](const float* s, _Float16* d) {
    f32_to_f16_t_kernel<<<dim3((MAT + 255) / 256), blk256, 0, stream>>>(s, d);
  };

  f32_to_f16_kernel<<<dim3((SEQ * HID + 255) / 256), blk256, 0, stream>>>(x_f32, xh, SEQ * HID);
  cvtT(Wout_f, Wout_h);
  for (int j = 0; j < 4; ++j) cvtT(fW[j], fWh[j]);
  for (int i = 0; i < 4; ++i) for (int j = 0; j < 5; ++j) cvtT(sW[i][j], sWh[i][j]);

  auto gemm16 = [&](const _Float16* A, long astride, const _Float16* Wt, const float* b,
                    _Float16* C, int M) {
    gemm512_wmma<0><<<dim3(8, M / 64), dim3(128), 0, stream>>>(A, astride, Wt, b, (void*)C, M);
  };
  auto gemm16t = [&](const _Float16* A, long astride, const _Float16* Wt, const float* b,
                     _Float16* Ct, int M) {
    gemm512_wmma<1><<<dim3(8, M / 64), dim3(128), 0, stream>>>(A, astride, Wt, b, (void*)Ct, M);
  };
  auto gemm32 = [&](const _Float16* A, long astride, const _Float16* Wt, const float* b,
                    float* C, int M) {
    gemm512_wmma<2><<<dim3(8, M / 64), dim3(128), 0, stream>>>(A, astride, Wt, b, (void*)C, M);
  };

  for (int i = 0; i < 4; ++i) {
    const int scale = 1 << i;
    const int L = SEQ / scale;
    const long astride = (long)HID * scale;            // x[::scale]
    gemm16 (xh, astride, sWh[i][0], sb[i][0], Qh, L);
    gemm16 (xh, astride, sWh[i][1], sb[i][1], Kh, L);
    gemm16t(xh, astride, sWh[i][2], sb[i][2], Vt, L);  // V written transposed
    flash_attn_wmma<<<dim3(NHEAD, L / 16), dim3(32), 0, stream>>>(Qh, Kh, Vt, AOh, L);
    gemm16 (AOh, HID, sWh[i][3], sb[i][3], Yh, L);     // Wo
    gemm16 (Yh,  HID, sWh[i][4], sb[i][4], Ph, L);     // Wp
    replicate_rows_kernel<<<dim3((SEQ * HID + 255) / 256), blk256, 0, stream>>>(Ph, stacked,
                                                                                scale, i);
  }

  gemm16(stacked, HID, fWh[0], fb[0], Qf, SEQ * 4);
  gemm16(stacked, HID, fWh[1], fb[1], Kf, SEQ * 4);
  gemm16(stacked, HID, fWh[2], fb[2], Vf, SEQ * 4);
  fusion_attn_kernel<<<dim3((SEQ * NHEAD * 4 + 255) / 256), blk256, 0, stream>>>(Qf, Kf, Vf, AOf);

  mean4_kernel<<<dim3((SEQ * HID + 255) / 256), blk256, 0, stream>>>(AOf, meanAO);
  gemm16(meanAO, HID, fWh[3], fb[3], T1, SEQ);         // fusion Wo + bo
  gemm32(T1, HID, Wout_h, bout, (float*)d_out, SEQ);   // Wout + bout -> f32 out
}
